// FusedLinearCrossEntropy_3985729651562
// MI455X (gfx1250) — matrix operand order repaired and checked by reading, not compiled, not run on hardware
//
#include <hip/hip_runtime.h>
#include <stdint.h>

typedef __attribute__((ext_vector_type(16))) __bf16 v16bf;
typedef __attribute__((ext_vector_type(8)))  float  v8f;

static constexpr int T_TOK   = 4096;
static constexpr int D_DIM   = 2048;
static constexpr int V_VOCAB = 32000;
static constexpr int NSLICES = 5;                       // vocab slices (grid.y)
static constexpr int TOK_PER_WG = 64;                   // 4 M-tiles of 16
static constexpr int NBLK_COLS  = 256;                  // cols per WG iteration (16 N-tiles)
static constexpr int COLS_PER_SLICE = V_VOCAB / NSLICES;            // 6400
static constexpr int NBLKS_PER_SLICE = COLS_PER_SLICE / NBLK_COLS;  // 25
static constexpr int LDA = D_DIM + 8;                   // padded LDS row stride (elements)
static constexpr int NPART = NSLICES * 4;               // 20 partials per token
static constexpr float LOG2E = 1.4426950408889634f;
static constexpr float LN2   = 0.6931471805599453f;

static_assert(NSLICES * NBLKS_PER_SLICE * NBLK_COLS == V_VOCAB, "vocab tiling");

__device__ __forceinline__ unsigned int f2bf(float f) {
  unsigned int u = __float_as_uint(f);
  u += 0x7FFFu + ((u >> 16) & 1u);     // round-to-nearest-even
  return u >> 16;
}
__device__ __forceinline__ float ex2(float x) { return __builtin_amdgcn_exp2f(x); }

// ---------------- kernel 0: zero the scalar output ----------------
__global__ void flce_zero(float* out) { out[0] = 0.0f; }

// ---------------- kernel 1: W f32 -> bf16 into workspace (fits L2) ----------------
__global__ __launch_bounds__(256) void flce_wcvt(const float* __restrict__ w,
                                                 unsigned short* __restrict__ w16) {
  long long i = ((long long)blockIdx.x * blockDim.x + threadIdx.x) * 8;
  if (i >= (long long)V_VOCAB * D_DIM) return;
  const float4* src = (const float4*)(w + i);
  float4 a = src[0], b = src[1];
  uint4 o;
  o.x = f2bf(a.x) | (f2bf(a.y) << 16);
  o.y = f2bf(a.z) | (f2bf(a.w) << 16);
  o.z = f2bf(b.x) | (f2bf(b.y) << 16);
  o.w = f2bf(b.z) | (f2bf(b.w) << 16);
  *(uint4*)(w16 + i) = o;
}

// ---------------- kernel 2: tiled bf16 WMMA GEMM + online logsumexp ----------------
__global__ __launch_bounds__(256) void flce_main(const float* __restrict__ hidden,
                                                 const int* __restrict__ labels,
                                                 const unsigned short* __restrict__ w16,
                                                 float* __restrict__ part) {
  extern __shared__ unsigned short sA[];   // [TOK_PER_WG][LDA] bf16, row-padded
  const int tid   = threadIdx.x;
  const int mb    = blockIdx.x;            // token block
  const int slice = blockIdx.y;            // vocab slice
  const int tok0  = mb * TOK_PER_WG;

  // Stage A tile: f32 global -> bf16 LDS. One row per pass, 8 elems/thread.
  for (int r = 0; r < TOK_PER_WG; ++r) {
    const float4* src = (const float4*)(hidden + (long long)(tok0 + r) * D_DIM + tid * 8);
    float4 a = src[0], b = src[1];
    uint4 o;
    o.x = f2bf(a.x) | (f2bf(a.y) << 16);
    o.y = f2bf(a.z) | (f2bf(a.w) << 16);
    o.z = f2bf(b.x) | (f2bf(b.y) << 16);
    o.w = f2bf(b.z) | (f2bf(b.w) << 16);
    *(uint4*)(sA + r * LDA + tid * 8) = o;
  }
  __syncthreads();

  const int lane = tid & 31;
  const int wave = tid >> 5;
  const int Mw   = wave & 1;               // 2 M-waves
  const int Nw   = wave >> 1;              // 4 N-waves
  const int mbase = Mw * 32;               // each wave covers 2 M-tiles (32 rows)
  const int l15   = lane & 15;
  const int hi    = (lane >= 16) ? 1 : 0;
  const int hoffA = hi ? 8 : 0;            // A: K interleave per ISA table
  const int hoffB = hi ? 8 : 0;            // B: K interleave per ISA table
  const int rowoff = hi ? 8 : 0;           // C/D row layout: lanes16-31 -> M+8

  // Labels for this wave's 16 (per-lane) row slots: 2 mtiles x 8 vgpr-rows
  int lab[2][8];
#pragma unroll
  for (int mi = 0; mi < 2; ++mi)
#pragma unroll
    for (int j = 0; j < 8; ++j)
      lab[mi][j] = labels[tok0 + mbase + mi * 16 + j + rowoff];

  float m[2][8], s[2][8], tg[2][8];
#pragma unroll
  for (int mi = 0; mi < 2; ++mi)
#pragma unroll
    for (int j = 0; j < 8; ++j) { m[mi][j] = -1e30f; s[mi][j] = 0.0f; tg[mi][j] = 0.0f; }

  int abase[2];
#pragma unroll
  for (int mi = 0; mi < 2; ++mi)
    abase[mi] = (mbase + mi * 16 + l15) * LDA + hoffA;   // A frag: row = lane&15

  for (int nblk = 0; nblk < NBLKS_PER_SLICE; ++nblk) {
    const int nb = slice * COLS_PER_SLICE + nblk * NBLK_COLS;
    const unsigned short* bptr[4];
    int coln[4];
#pragma unroll
    for (int ni = 0; ni < 4; ++ni) {
      coln[ni] = nb + Nw * 64 + ni * 16 + l15;                       // B frag: col = lane&15
      bptr[ni] = w16 + (long long)coln[ni] * D_DIM + hoffB;
    }

    v8f C[2][4];
#pragma unroll
    for (int mi = 0; mi < 2; ++mi)
#pragma unroll
      for (int ni = 0; ni < 4; ++ni) C[mi][ni] = (v8f){0,0,0,0,0,0,0,0};

#pragma unroll 2
    for (int k0 = 0; k0 < D_DIM; k0 += 32) {
      union Frag { uint4 u[2]; v16bf v; } a[2], b[4];
#pragma unroll
      for (int mi = 0; mi < 2; ++mi) {
        const uint4* p = (const uint4*)(sA + abase[mi] + k0);
        a[mi].u[0] = p[0];        // K 0..7   (this half-wave's span)
        a[mi].u[1] = p[2];        // K 16..23 (+16 elements)
      }
#pragma unroll
      for (int ni = 0; ni < 4; ++ni) {
        const uint4* p = (const uint4*)(bptr[ni] + k0);
        b[ni].u[0] = p[0];        // K 0..7   (this half-wave's span)
        b[ni].u[1] = p[2];        // K 16..23 (+16 elements)
      }
#pragma unroll
      for (int mi = 0; mi < 2; ++mi)
#pragma unroll
        for (int ni = 0; ni < 4; ++ni)
          C[mi][ni] = __builtin_amdgcn_wmma_f32_16x16x32_bf16(
              false, a[mi].v, false, b[ni].v, (short)0, C[mi][ni], false, false);
    }

    // Online logsumexp update (base-2 domain) + target-logit capture
#pragma unroll
    for (int mi = 0; mi < 2; ++mi)
#pragma unroll
      for (int ni = 0; ni < 4; ++ni)
#pragma unroll
        for (int j = 0; j < 8; ++j) {
          float c = C[mi][ni][j];
          tg[mi][j] += (lab[mi][j] == coln[ni]) ? c : 0.0f;
          float c2 = c * LOG2E;
          float mo = m[mi][j];
          float mn = fmaxf(mo, c2);
          s[mi][j] = s[mi][j] * ex2(mo - mn) + ex2(c2 - mn);
          m[mi][j] = mn;
        }
  }

  // Column reduction across the 16 lanes of each half-wave, then write partials.
#pragma unroll
  for (int mi = 0; mi < 2; ++mi)
#pragma unroll
    for (int j = 0; j < 8; ++j) {
      float mm = m[mi][j], ss = s[mi][j], tt = tg[mi][j];
#pragma unroll
      for (int off = 1; off < 16; off <<= 1) {   // xor<16 stays inside each half
        float mo = __shfl_xor(mm, off, 32);
        float so = __shfl_xor(ss, off, 32);
        float to = __shfl_xor(tt, off, 32);
        float mn = fmaxf(mm, mo);
        ss = ss * ex2(mm - mn) + so * ex2(mo - mn);
        mm = mn;
        tt += to;
      }
      if (l15 == j) {
        int t = tok0 + mbase + mi * 16 + j + rowoff;
        float* o = part + ((long long)t * NPART + slice * 4 + Nw) * 3;
        o[0] = mm; o[1] = ss; o[2] = tt;
      }
    }
}

// ---------------- kernel 3: combine partials, reduce, atomic add ----------------
__global__ __launch_bounds__(256) void flce_reduce(const float* __restrict__ part,
                                                   const int* __restrict__ labels,
                                                   const int* __restrict__ ntok,
                                                   float* __restrict__ out) {
  __shared__ float red[256];
  const int t = blockIdx.x * 256 + threadIdx.x;
  const float* p = part + (long long)t * NPART * 3;
  float M = -1e30f, S = 0.0f, tgt = 0.0f;
#pragma unroll
  for (int i = 0; i < NPART; ++i) {
    float mi = p[i * 3 + 0];
    float si = p[i * 3 + 1];
    tgt += p[i * 3 + 2];
    float Mn = fmaxf(M, mi);
    S = S * ex2(M - Mn) + si * ex2(mi - Mn);
    M = Mn;
  }
  float lse = (M + __builtin_amdgcn_logf(S)) * LN2;   // v_log_f32 is log2
  float pt  = (labels[t] == -100) ? 0.0f : (lse - tgt);
  red[threadIdx.x] = pt;
  __syncthreads();
  for (int w = 128; w > 0; w >>= 1) {
    if (threadIdx.x < w) red[threadIdx.x] += red[threadIdx.x + w];
    __syncthreads();
  }
  if (threadIdx.x == 0) atomicAdd(out, red[0] / (float)ntok[0]);
}

extern "C" void kernel_launch(void* const* d_in, const int* in_sizes, int n_in,
                              void* d_out, int out_size, void* d_ws, size_t ws_size,
                              hipStream_t stream) {
  (void)in_sizes; (void)n_in; (void)out_size; (void)ws_size;
  const float* hidden = (const float*)d_in[0];
  const int*   labels = (const int*)d_in[1];
  const float* weight = (const float*)d_in[2];
  const int*   ntok   = (const int*)d_in[3];
  float* out = (float*)d_out;

  unsigned short* w16 = (unsigned short*)d_ws;                       // 131,072,000 B (fits L2)
  float* part = (float*)((char*)d_ws + (size_t)V_VOCAB * D_DIM * 2); // T*20*3 floats

  flce_zero<<<1, 1, 0, stream>>>(out);

  const long long nvec = (long long)V_VOCAB * D_DIM / 8;             // 8,192,000
  flce_wcvt<<<(int)((nvec + 255) / 256), 256, 0, stream>>>(weight, w16);

  dim3 grid(T_TOK / TOK_PER_WG, NSLICES);                            // 64 x 5
  size_t lds = (size_t)TOK_PER_WG * LDA * sizeof(unsigned short);    // 263,168 B < 320 KB
  flce_main<<<grid, 256, lds, stream>>>(hidden, labels, w16, part);

  flce_reduce<<<T_TOK / 256, 256, 0, stream>>>(part, labels, ntok, out);
}
